// HungarianMatcher_48344151884117
// MI455X (gfx1250) — matrix operand order repaired and checked by reading, not compiled, not run on hardware
//
#include <hip/hip_runtime.h>
#include <hip/hip_bf16.h>
#include <math.h>

#define BSZ   16
#define NQ    900
#define NCLS  1000
#define TT    100
#define NTGT  (BSZ*TT)     // 1600
#define NROW  (BSZ*NQ)     // 14400
#define NCH   32           // ceil(1000/32) chunks of 32 classes (padded to 1024)

typedef __attribute__((ext_vector_type(16))) _Float16     v16h;
typedef __attribute__((ext_vector_type(8)))  float        v8f;
typedef __attribute__((ext_vector_type(8)))  unsigned int v8u;

// ---------------------------------------------------------------------------
// Kernel 1: cost matrix. One block per 16-row strip of the 14400-row output.
//   Cm[i,j] = sum_k |qbox[i,k]-tbox[j,k]| - softmax(logits[i])[tgt_ids[j]]
// Class term computed as a one-hot split-f16 WMMA GEMM (exact to fp32).
// ---------------------------------------------------------------------------
__global__ __launch_bounds__(256) void cost_kernel(
    const float* __restrict__ logits,   // [NROW, NCLS]
    const float* __restrict__ qbox,     // [NROW, 4]
    const int*   __restrict__ tids,     // [NTGT]
    const float* __restrict__ tbox,     // [NTGT, 4]
    float* __restrict__ Cm)             // [NROW, NTGT]
{
  // A-matrix fragments (16x32 f16 per chunk) laid out in WMMA order:
  // element (m,kk) of chunk c -> lane = m + 16*(grp&1), half h = kk - 8*((grp+1)>>1)
  __shared__ _Float16 Ahi[NCH*32*16];   // 32 KB
  __shared__ _Float16 Alo[NCH*32*16];   // 32 KB  (total 64 KB, CDNA5 WGP has 320 KB)

  const int strip = blockIdx.x;         // 0..899
  const int i0    = strip * 16;
  const int wave  = threadIdx.x >> 5;   // 8 waves (wave32)
  const int lane  = threadIdx.x & 31;

  // ---------------- Phase 0: softmax -> split-f16 fragments in LDS ----------
  for (int rr = 0; rr < 2; ++rr) {
    const int m = wave * 2 + rr;        // row within strip, 0..15
    const float* Lrow = logits + (size_t)(i0 + m) * NCLS;
    float vals[32];
    float mx = -INFINITY;
    #pragma unroll
    for (int t = 0; t < 32; ++t) {
      int k = lane + 32 * t;            // 0..1023 (pad beyond 1000)
      float x = (k < NCLS) ? Lrow[k] : -INFINITY;
      vals[t] = x;
      mx = fmaxf(mx, x);
    }
    #pragma unroll
    for (int off = 16; off >= 1; off >>= 1) mx = fmaxf(mx, __shfl_xor(mx, off, 32));
    float sum = 0.f;
    #pragma unroll
    for (int t = 0; t < 32; ++t) { float e = __expf(vals[t] - mx); vals[t] = e; sum += e; }
    #pragma unroll
    for (int off = 16; off >= 1; off >>= 1) sum += __shfl_xor(sum, off, 32);
    const float inv = 1.0f / sum;
    #pragma unroll
    for (int t = 0; t < 32; ++t) {
      int k = lane + 32 * t;
      float p = vals[t] * inv;          // pad classes -> exp(-inf)=0 -> p=0
      _Float16 hi = (_Float16)p;
      _Float16 lo = (_Float16)(p - (float)hi);
      int c   = k >> 5;
      int kk  = k & 31;
      int grp = kk >> 3;
      int dl  = m + ((grp & 1) << 4);
      int h   = kk - 8 * ((grp + 1) >> 1);
      int idx = ((c << 5) + dl) * 16 + h;
      Ahi[idx] = hi;
      Alo[idx] = lo;
    }
  }
  __syncthreads();

  // Per-lane query boxes for the C/D fragment rows this lane owns:
  // C/D layout: lane L covers column n=L&15, rows m = r + 8*(L>>4), r=0..7
  const int khrow = lane >> 4;
  const int n     = lane & 15;
  float qb[8][4];
  #pragma unroll
  for (int r = 0; r < 8; ++r) {
    int m = r + 8 * khrow;
    #pragma unroll
    for (int k = 0; k < 4; ++k) qb[r][k] = qbox[(size_t)(i0 + m) * 4 + k];
  }

  // ---------------- Phase 1: per-wave target tiles ---------------------------
  for (int tt = wave; tt < NTGT / 16; tt += 8) {
    const int j0   = tt * 16;
    const int myid = tids[j0 + n];      // class id for this lane's column (dup in hi half)

    // Prefetch next tile's target data (lowers to global_prefetch_b8)
    if (tt + 8 < NTGT / 16) {
      __builtin_prefetch(tbox + (size_t)(j0 + 128 + n) * 4, 0, 3);
      __builtin_prefetch(tids + (j0 + 128 + n), 0, 3);
    }

    // Live-chunk mask over the 16 ids of this tile; force wave-uniform so the
    // chunk loop is SALU-controlled and EXEC stays all-1s around the WMMAs.
    unsigned mask = 1u << (myid >> 5);
    #pragma unroll
    for (int off = 16; off >= 1; off >>= 1) mask |= __shfl_xor(mask, off, 32);
    unsigned smask = (unsigned)__builtin_amdgcn_readfirstlane((int)mask);

    // bbox L1 cost -> f32 accumulator tile
    float tb[4];
    #pragma unroll
    for (int k = 0; k < 4; ++k) tb[k] = tbox[(size_t)(j0 + n) * 4 + k];
    v8f C;
    #pragma unroll
    for (int r = 0; r < 8; ++r) {
      float s = 0.f;
      #pragma unroll
      for (int k = 0; k < 4; ++k) s += fabsf(qb[r][k] - tb[k]);
      C[r] = s;
    }

    // Accumulate -prob via one-hot split-f16 WMMA, skipping dead K-chunks
    while (smask) {
      const int c = __ffs(smask) - 1;          // scalar (s_ff1)
      smask &= smask - 1;
      const int local = myid - (c << 5);       // 0..31 if this column hits chunk c
      const int s     = (((local >> 3) & 1) == khrow) ? local - 8 * (((local >> 3) + 1) >> 1) : -1;  // position within this lane's K half (table order)
      // B fragment (32x16 f16): lane = n + 16*khalf; VGPR r holds K halves (2r,2r+1)
      v8u bb = {};
      #pragma unroll
      for (int r = 0; r < 8; ++r) {
        unsigned w = (s == 2 * r)     ? 0x0000BC00u : 0u;   // f16 -1.0 in low half
        w         |= (s == 2 * r + 1) ? 0xBC000000u : 0u;   // f16 -1.0 in high half
        bb[r] = w;
      }
      const v16h B = __builtin_bit_cast(v16h, bb);

      const v16h a_hi = *(const v16h*)(&Ahi[((c << 5) + lane) * 16]);
      C = __builtin_amdgcn_wmma_f32_16x16x32_f16(false, a_hi, false, B, (short)0, C, false, false);
      const v16h a_lo = *(const v16h*)(&Alo[((c << 5) + lane) * 16]);
      C = __builtin_amdgcn_wmma_f32_16x16x32_f16(false, a_lo, false, B, (short)0, C, false, false);
    }

    // Store tile (16 consecutive floats per (row,half) -> 64B-coalesced)
    #pragma unroll
    for (int r = 0; r < 8; ++r) {
      int m = r + 8 * khrow;
      Cm[(size_t)(i0 + m) * NTGT + j0 + n] = C[r];
    }
  }
}

// ---------------------------------------------------------------------------
// Kernel 2: extract the per-batch 900x100 block diagonal, transposed to
// costT[b][t][q] (contiguous rows for the JV scans).
// ---------------------------------------------------------------------------
__global__ __launch_bounds__(256) void transpose_kernel(
    const float* __restrict__ Cm, float* __restrict__ costT)
{
  const int b = blockIdx.x;
  for (int idx = threadIdx.x; idx < NQ * TT; idx += 256) {
    int q = idx / TT, t = idx % TT;
    costT[(size_t)b * TT * NQ + (size_t)t * NQ + q] =
        Cm[(size_t)(b * NQ + q) * NTGT + b * TT + t];
  }
}

// ---------------------------------------------------------------------------
// Kernel 3: Jonker-Volgenant shortest augmenting path, one block per batch.
// Double precision to match the numpy float64 reference (tie fidelity).
// ---------------------------------------------------------------------------
__global__ __launch_bounds__(256) void lsa_kernel(
    const float* __restrict__ costT, float* __restrict__ idx_i, float* __restrict__ idx_j)
{
  const int b = blockIdx.x;
  const float* cost = costT + (size_t)b * TT * NQ;   // [TT][NQ]

  __shared__ double u[TT + 1];
  __shared__ double v[NQ + 1];
  __shared__ double minv[NQ + 1];
  __shared__ short  way[NQ + 1];
  __shared__ short  p[NQ + 1];
  __shared__ unsigned char used[NQ + 1];
  __shared__ double redv[256];
  __shared__ int    redj[256];
  __shared__ int    cols[TT];

  const int tidx = threadIdx.x;
  for (int j = tidx; j <= NQ; j += 256) { v[j] = 0.0; p[j] = 0; way[j] = 0; }
  for (int i = tidx; i <= TT; i += 256) u[i] = 0.0;
  __syncthreads();

  for (int i = 1; i <= TT; ++i) {
    if (tidx == 0) p[0] = (short)i;
    for (int j = tidx; j <= NQ; j += 256) { minv[j] = 1e300; used[j] = 0; }
    __syncthreads();

    int j0 = 0;   // kept uniform across all threads
    while (true) {
      if (tidx == 0) used[j0] = 1;
      __syncthreads();
      const int    i0  = (int)p[j0];
      const double ui0 = u[i0];

      // parallel scan over free columns + local argmin (tie -> smallest j)
      double bv = 1e301; int bj = NQ + 1;
      for (int j = tidx + 1; j <= NQ; j += 256) {
        if (!used[j]) {
          double cur = (double)cost[(size_t)(i0 - 1) * NQ + (j - 1)] - ui0 - v[j];
          double mj  = minv[j];
          if (cur < mj) { minv[j] = cur; way[j] = (short)j0; mj = cur; }
          if (mj < bv || (mj == bv && j < bj)) { bv = mj; bj = j; }
        }
      }
      redv[tidx] = bv; redj[tidx] = bj;
      __syncthreads();
      for (int s = 128; s >= 1; s >>= 1) {
        if (tidx < s) {
          double ov = redv[tidx + s]; int oj = redj[tidx + s];
          if (ov < redv[tidx] || (ov == redv[tidx] && oj < redj[tidx])) {
            redv[tidx] = ov; redj[tidx] = oj;
          }
        }
        __syncthreads();
      }
      const int    j1    = redj[0];
      const double delta = redv[0];
      __syncthreads();

      // dual update (rows on the alternating tree are distinct -> no races)
      for (int j = tidx; j <= NQ; j += 256) {
        if (used[j]) { u[p[j]] += delta; v[j] -= delta; }
        else         { minv[j] -= delta; }
      }
      __syncthreads();

      j0 = j1;
      if (p[j0] == 0) break;
    }

    // augment along the path (serial, thread 0)
    if (tidx == 0) {
      int jj = j0;
      while (jj) { int jn = way[jj]; p[jj] = p[jn]; jj = jn; }
    }
    __syncthreads();
  }

  // cols[t] = matched query for target t
  for (int j = tidx + 1; j <= NQ; j += 256) {
    int pi = (int)p[j];
    if (pi > 0) cols[pi - 1] = j - 1;
  }
  __syncthreads();

  // idx_i = sorted(cols), idx_j = argsort(cols)  (cols are distinct)
  for (int t = tidx; t < TT; t += 256) {
    int c = cols[t], rank = 0;
    for (int t2 = 0; t2 < TT; ++t2) rank += (cols[t2] < c);
    idx_i[b * TT + rank] = (float)c;
    idx_j[b * TT + rank] = (float)t;
  }
}

// ---------------------------------------------------------------------------
extern "C" void kernel_launch(void* const* d_in, const int* in_sizes, int n_in,
                              void* d_out, int out_size, void* d_ws, size_t ws_size,
                              hipStream_t stream)
{
  const float* logits = (const float*)d_in[0];   // pred_logits [16,900,1000]
  const float* qboxes = (const float*)d_in[1];   // pred_boxes  [16,900,4]
  const int*   tids   = (const int*)  d_in[2];   // tgt_ids     [1600]
  const float* tboxes = (const float*)d_in[3];   // tgt_bbox    [1600,4]

  float* Cm    = (float*)d_out;                       // 23,040,000 floats
  float* idx_i = Cm + (size_t)NROW * NTGT;            // 1600 floats
  float* idx_j = idx_i + NTGT;                        // 1600 floats
  float* costT = (float*)d_ws;                        // 16*100*900 floats = 5.76 MB

  cost_kernel<<<NROW / 16, 256, 0, stream>>>(logits, qboxes, tids, tboxes, Cm);
  transpose_kernel<<<BSZ, 256, 0, stream>>>(Cm, costT);
  lsa_kernel<<<BSZ, 256, 0, stream>>>(costT, idx_i, idx_j);
}
